// TransformerMT_27779848471159
// MI455X (gfx1250) — matrix operand order repaired and checked by reading, not compiled, not run on hardware
//
#include <hip/hip_runtime.h>
#include <hip/hip_bf16.h>

typedef __attribute__((ext_vector_type(16))) _Float16 v16h;
typedef __attribute__((ext_vector_type(8)))  float    v8f;

union H16 { v16h v; _Float16 h[16]; uint4 q[2]; };

__device__ __forceinline__ v8f wmma_f16(const H16& a, const H16& b, v8f c) {
    return __builtin_amdgcn_wmma_f32_16x16x32_f16(
        /*neg_a=*/false, a.v, /*neg_b=*/false, b.v,
        /*c_mod=*/(short)0, c, /*reuse_a=*/false, /*reuse_b=*/false);
}

// convert 8 consecutive fp32 (16B-aligned) to 8 f16 at h[0..7]
__device__ __forceinline__ void cvt8(_Float16* h, const float* p) {
    float4 f0 = *(const float4*)p;
    float4 f1 = *(const float4*)(p + 4);
    h[0] = (_Float16)f0.x; h[1] = (_Float16)f0.y; h[2] = (_Float16)f0.z; h[3] = (_Float16)f0.w;
    h[4] = (_Float16)f1.x; h[5] = (_Float16)f1.y; h[6] = (_Float16)f1.z; h[7] = (_Float16)f1.w;
}

// ---------------------------------------------------------------------------
// Generic GEMM: C[M,N] = act(A[M,K] @ W[K,N] + bias),  fp32 in/out, f16 WMMA.
// Block tile 128x128, BK=32, 256 threads = 8 waves; wave w -> rows [16w,16w+16),
// 8 accumulators covering the 128 output columns.
// Double-buffered LDS: next K-tile's global loads are issued before the WMMA
// block and retired after it, so load latency hides behind matrix ops.
// Grid: x = M-block (fast) so consecutive blocks share one W column slice
// (W read once from HBM; A slice stays L2-resident across N-stripes).
// ---------------------------------------------------------------------------
#define BM 128
#define BN 128
#define BK 32
#define LDK 40   // padded K-stride in halfs (80B, 16B-aligned)

__global__ __launch_bounds__(256) void gemm_f16w(
    const float* __restrict__ A, const float* __restrict__ W,
    const float* __restrict__ bias, float* __restrict__ C,
    int M, int N, int K, int relu)
{
    __shared__ __align__(16) _Float16 As[2][BM][LDK];
    __shared__ __align__(16) _Float16 Bt[2][BN][LDK];   // transposed: [n][k]

    const int tid  = threadIdx.x;
    const int lane = tid & 31;
    const int w    = tid >> 5;
    const int m16  = lane & 15;
    const int hi   = lane >> 4;
    const int rowBase = blockIdx.x * BM;   // M-block varies fastest
    const int nBase   = blockIdx.y * BN;

    float4 fa[4];      // staged A tile (16 fp32/thread)
    float  fb[16];     // staged W tile (16 fp32/thread)

    auto load_tile = [&](int k0) {
#pragma unroll
        for (int t = 0; t < 4; ++t) {
            int linear = t * 1024 + tid * 4;
            int r = linear >> 5;
            int c = linear & 31;
            fa[t] = *(const float4*)(A + (size_t)(rowBase + r) * K + k0 + c);
        }
#pragma unroll
        for (int t = 0; t < 16; ++t) {
            int linear = t * 256 + tid;
            int kk = linear >> 7;
            int n  = linear & 127;
            fb[t] = W[(size_t)(k0 + kk) * N + nBase + n];
        }
    };
    auto store_tile = [&](int buf) {
#pragma unroll
        for (int t = 0; t < 4; ++t) {
            int linear = t * 1024 + tid * 4;
            int r = linear >> 5;
            int c = linear & 31;
            As[buf][r][c + 0] = (_Float16)fa[t].x; As[buf][r][c + 1] = (_Float16)fa[t].y;
            As[buf][r][c + 2] = (_Float16)fa[t].z; As[buf][r][c + 3] = (_Float16)fa[t].w;
        }
#pragma unroll
        for (int t = 0; t < 16; ++t) {
            int linear = t * 256 + tid;
            int kk = linear >> 7;
            int n  = linear & 127;
            Bt[buf][n][kk] = (_Float16)fb[t];
        }
    };

    v8f acc[8];
#pragma unroll
    for (int g = 0; g < 8; ++g) acc[g] = v8f{0,0,0,0,0,0,0,0};

    // prologue: stage tile 0
    load_tile(0);
    store_tile(0);
    __syncthreads();

    const int nsteps = K / BK;
    for (int step = 0; step < nsteps; ++step) {
        const int cur = step & 1;

        // issue next tile's global loads (retired after the WMMA block)
        if (step + 1 < nsteps) {
            load_tile((step + 1) * BK);
            int k1 = (step + 1) * BK;
            if (k1 + BK < K) {   // prefetch tile after next
                int pr = tid >> 1;
                __builtin_prefetch(A + (size_t)(rowBase + pr) * K + k1 + BK + (tid & 1) * 16, 0, 1);
                __builtin_prefetch(W + (size_t)(k1 + BK + (tid & 31)) * N + nBase + (tid >> 5) * 16, 0, 1);
            }
        }

        // A fragment (16x32 f16 layout): lanes 0-15 -> K{0..7,16..23}, 16-31 -> K{8..15,24..31}
        H16 a;
        {
            int m   = w * 16 + m16;
            int klo = hi ? 8 : 0;
            a.q[0] = *(const uint4*)&As[cur][m][klo];
            a.q[1] = *(const uint4*)&As[cur][m][klo + 16];
        }
#pragma unroll
        for (int g = 0; g < 8; ++g) {
            // B fragment (32x16): lane n = col, lanes 0-15 -> K{0..7,16..23}, 16-31 -> K{8..15,24..31}
            H16 b;
            int n  = g * 16 + m16;
            int kb = hi ? 8 : 0;
            b.q[0] = *(const uint4*)&Bt[cur][n][kb];
            b.q[1] = *(const uint4*)&Bt[cur][n][kb + 16];
            acc[g] = wmma_f16(a, b, acc[g]);
        }

        // convert + store the staged tile into the other buffer
        if (step + 1 < nsteps) store_tile(cur ^ 1);
        __syncthreads();
    }

    // epilogue: C layout VGPR r <-> M = r + 8*hi, lane&15 <-> N
#pragma unroll
    for (int g = 0; g < 8; ++g) {
        int col = nBase + g * 16 + m16;
        float bv = bias[col];
#pragma unroll
        for (int r = 0; r < 8; ++r) {
            int row = rowBase + w * 16 + r + hi * 8;
            float v = acc[g][r] + bv;
            if (relu) v = fmaxf(v, 0.0f);
            C[(size_t)row * N + col] = v;
        }
    }
}

// ---------------------------------------------------------------------------
// Fused multi-head attention (flash-style streaming softmax).
// Grid: (B*H, Sq/128); 256 threads = 8 waves; wave -> one 16-query tile.
// Q,K,V,O are [B, S, 1024] fp32 with head h at column offset h*64.
// mode 0: key-pad mask (toks[b,key] != 0)    -- encoder self & cross attn
// mode 1: causal & query-pad (toks[b,q]!=0 && q>=key) -- decoder self attn
//         (key loop stops at the block's last query: uniform early exit)
// ---------------------------------------------------------------------------
__global__ __launch_bounds__(256) void attention_kernel(
    const float* __restrict__ Q, const float* __restrict__ Kb,
    const float* __restrict__ Vb, float* __restrict__ O,
    const int* __restrict__ toks, int Sq, int Sk, int mode)
{
    __shared__ __align__(16) _Float16 Pl[8][16][16];   // per-wave P tile
    __shared__ __align__(16) _Float16 Vt[8][64][16];   // per-wave V tile, [hd][key]

    const int tid  = threadIdx.x;
    const int lane = tid & 31;
    const int w    = tid >> 5;
    const int m16  = lane & 15;
    const int hi   = lane >> 4;
    const int bh   = blockIdx.x;
    const int b    = bh >> 4;          // H = 16
    const int h    = bh & 15;
    const int qbase = (blockIdx.y * 8 + w) * 16;
    const int D = 1024, HD = 64;

    // Q fragments: a0 covers hd 0..31, a1 covers hd 32..63
    H16 a0, a1;
    {
        const float* qrow = Q + ((size_t)b * Sq + qbase + m16) * D + h * HD;
        int klo = hi ? 8 : 0;
        cvt8(a0.h,     qrow + klo);
        cvt8(a0.h + 8, qrow + klo + 16);
        cvt8(a1.h,     qrow + klo + 32);
        cvt8(a1.h + 8, qrow + klo + 48);
    }

    float mrow[8], lrow[8];
    bool  qok[8];
    v8f o[4];
#pragma unroll
    for (int r = 0; r < 8; ++r) {
        mrow[r] = -1e30f; lrow[r] = 0.0f;
        int qi = qbase + r + hi * 8;
        qok[r] = (mode == 1) ? (toks[(size_t)b * Sq + qi] != 0) : true;
    }
#pragma unroll
    for (int g = 0; g < 4; ++g) o[g] = v8f{0,0,0,0,0,0,0,0};

    const float scale = 0.125f;   // 1/sqrt(64)

    // causal: keys beyond the block's last query are masked for every wave
    int jend = (mode == 1) ? (int)((blockIdx.y + 1) * 128) : Sk;
    if (jend > Sk) jend = Sk;

    for (int j = 0; j < jend; j += 16) {
        // stage V tile [16 keys x 64 hd] transposed into Vt[hd][key]:
        // issue all 8 global loads, then convert+store
        float4 fv[8];
#pragma unroll
        for (int t = 0; t < 8; ++t) {
            int idx = t * 32 + lane;          // 256 float4 units
            int vr  = idx >> 4;               // key row 0..15
            int c4  = (idx & 15) * 4;
            fv[t] = *(const float4*)(Vb + ((size_t)b * Sk + j + vr) * D + h * HD + c4);
        }
#pragma unroll
        for (int t = 0; t < 8; ++t) {
            int idx = t * 32 + lane;
            int vr  = idx >> 4;
            int c4  = (idx & 15) * 4;
            Vt[w][c4 + 0][vr] = (_Float16)fv[t].x;
            Vt[w][c4 + 1][vr] = (_Float16)fv[t].y;
            Vt[w][c4 + 2][vr] = (_Float16)fv[t].z;
            Vt[w][c4 + 3][vr] = (_Float16)fv[t].w;
        }

        // K^T fragments: B matrix is [hd x key]; lane = key col, lanes 0-15 -> hd{0..7,16..23}, 16-31 -> hd{8..15,24..31}
        H16 b0, b1;
        {
            const float* krow = Kb + ((size_t)b * Sk + j + m16) * D + h * HD;
            int kb = hi ? 8 : 0;
            cvt8(b0.h,     krow + kb);
            cvt8(b0.h + 8, krow + kb + 16);
            cvt8(b1.h,     krow + 32 + kb);
            cvt8(b1.h + 8, krow + 32 + kb + 16);
        }

        v8f s = v8f{0,0,0,0,0,0,0,0};
        s = wmma_f16(a0, b0, s);
        s = wmma_f16(a1, b1, s);

        const int key = j + m16;
        bool kok = true;
        if (mode == 0) kok = (toks[(size_t)b * Sk + key] != 0);

#pragma unroll
        for (int r = 0; r < 8; ++r) {
            int qi = qbase + r + hi * 8;
            float sv = s[r] * scale;
            bool ok = (mode == 1) ? (qok[r] && (qi >= key)) : kok;
            if (!ok) sv = -1e9f;
            // row max across 16 lanes of this half-wave
            float rm = sv;
            for (int off = 1; off < 16; off <<= 1) rm = fmaxf(rm, __shfl_xor(rm, off, 32));
            float mnew  = fmaxf(mrow[r], rm);
            float alpha = __expf(mrow[r] - mnew);
            float p     = __expf(sv - mnew);
            float rs = p;
            for (int off = 1; off < 16; off <<= 1) rs += __shfl_xor(rs, off, 32);
            lrow[r] = lrow[r] * alpha + rs;
            mrow[r] = mnew;
#pragma unroll
            for (int g = 0; g < 4; ++g) o[g][r] *= alpha;
            Pl[w][r + hi * 8][m16] = (_Float16)p;
        }
        __syncthreads();

        // P as A-fragment (K 0..15 live, K 16..31 zero)
        H16 pa;
        pa.q[0] = *(const uint4*)&Pl[w][m16][hi ? 8 : 0];
        pa.q[1] = make_uint4(0, 0, 0, 0);

#pragma unroll
        for (int g = 0; g < 4; ++g) {
            H16 bv;
            const _Float16* vp = &Vt[w][g * 16 + m16][hi ? 8 : 0];
            bv.q[0] = *(const uint4*)vp;
            bv.q[1] = make_uint4(0, 0, 0, 0);
            o[g] = wmma_f16(pa, bv, o[g]);
        }
        __syncthreads();
    }

    // write normalized output
#pragma unroll
    for (int g = 0; g < 4; ++g) {
#pragma unroll
        for (int r = 0; r < 8; ++r) {
            int qi = qbase + r + hi * 8;
            O[((size_t)b * Sq + qi) * D + h * HD + g * 16 + m16] = o[g][r] / lrow[r];
        }
    }
}

// ---------------------------------------------------------------------------
// Out = LayerNorm(X + R) * g + b     (row length 1024, population variance)
// ---------------------------------------------------------------------------
__global__ __launch_bounds__(256) void ln_residual_kernel(
    const float* __restrict__ X, const float* __restrict__ R,
    const float* __restrict__ G, const float* __restrict__ Bv,
    float* __restrict__ Out)
{
    __shared__ float red1[256];
    __shared__ float red2[256];
    const int row = blockIdx.x;
    const int tid = threadIdx.x;
    const float* x = X + (size_t)row * 1024;
    const float* r = R + (size_t)row * 1024;
    float v[4], s = 0.f, s2 = 0.f;
#pragma unroll
    for (int i = 0; i < 4; ++i) {
        int d = tid * 4 + i;
        v[i] = x[d] + r[d];
        s += v[i]; s2 += v[i] * v[i];
    }
    red1[tid] = s; red2[tid] = s2;
    __syncthreads();
    for (int o2 = 128; o2 > 0; o2 >>= 1) {
        if (tid < o2) { red1[tid] += red1[tid + o2]; red2[tid] += red2[tid + o2]; }
        __syncthreads();
    }
    float mean = red1[0] * (1.0f / 1024.0f);
    float var  = red2[0] * (1.0f / 1024.0f) - mean * mean;
    float inv  = rsqrtf(var + 1e-5f);
    float* out = Out + (size_t)row * 1024;
#pragma unroll
    for (int i = 0; i < 4; ++i) {
        int d = tid * 4 + i;
        out[d] = (v[i] - mean) * inv * G[d] + Bv[d];
    }
}

// ---------------------------------------------------------------------------
// Out[row] = emb[tok[row]] + positional encoding(row % S)
// ---------------------------------------------------------------------------
__global__ __launch_bounds__(256) void embed_pe_kernel(
    const int* __restrict__ toks, const float* __restrict__ emb,
    float* __restrict__ Out, int S)
{
    const int row = blockIdx.x;
    const int s   = row % S;
    const int tid = threadIdx.x;
    const int t   = toks[row];
    const float* e = emb + (size_t)t * 1024;
    float* o = Out + (size_t)row * 1024;
#pragma unroll
    for (int i = 0; i < 4; ++i) {
        int d = tid * 4 + i;
        float div = __expf(-logf(10000.0f) * (float)(d & ~1) * (1.0f / 1024.0f));
        float ang = (float)s * div;
        o[d] = e[d] + ((d & 1) ? __cosf(ang) : __sinf(ang));
    }
}

// ---------------------------------------------------------------------------
// Host orchestration
// ---------------------------------------------------------------------------
extern "C" void kernel_launch(void* const* d_in, const int* in_sizes, int n_in,
                              void* d_out, int out_size, void* d_ws, size_t ws_size,
                              hipStream_t stream) {
    (void)in_sizes; (void)n_in; (void)out_size; (void)ws_size;

    const int*   src       = (const int*)  d_in[0];
    const int*   trg       = (const int*)  d_in[1];
    const float* src_emb   = (const float*)d_in[2];
    const float* trg_emb   = (const float*)d_in[3];
    const float* enc_attn_w= (const float*)d_in[4];
    const float* enc_attn_b= (const float*)d_in[5];
    const float* enc_ff1_w = (const float*)d_in[6];
    const float* enc_ff1_b = (const float*)d_in[7];
    const float* enc_ff2_w = (const float*)d_in[8];
    const float* enc_ff2_b = (const float*)d_in[9];
    const float* enc_ln_g  = (const float*)d_in[10];
    const float* enc_ln_b  = (const float*)d_in[11];
    const float* dec_self_w= (const float*)d_in[12];
    const float* dec_self_b= (const float*)d_in[13];
    const float* dec_cross_w=(const float*)d_in[14];
    const float* dec_cross_b=(const float*)d_in[15];
    const float* dec_ff1_w = (const float*)d_in[16];
    const float* dec_ff1_b = (const float*)d_in[17];
    const float* dec_ff2_w = (const float*)d_in[18];
    const float* dec_ff2_b = (const float*)d_in[19];
    const float* dec_ln_g  = (const float*)d_in[20];
    const float* dec_ln_b  = (const float*)d_in[21];
    const float* out_w     = (const float*)d_in[22];
    const float* out_b     = (const float*)d_in[23];

    const int L = 6, Dm = 1024, F = 4096, M = 4096 /* B*S */, S = 1024;
    const size_t MD = (size_t)M * Dm;            // 4M floats
    const size_t WD = (size_t)Dm * Dm;           // 1M floats (one DxD weight)

    float* ws  = (float*)d_ws;
    float* x   = ws;             // encoder stream / encoder output
    float* y   = x  + MD;        // decoder stream
    float* t   = y  + MD;        // residual branch output
    float* Qb  = t  + MD;
    float* Kf  = Qb + MD;
    float* Vf  = Kf + MD;
    float* aO  = Vf + MD;        // concat-head attention output
    float* ffh = aO + MD;        // [M, F] hidden

    auto gemm = [&](const float* A, const float* W, const float* bias, float* C,
                    int m, int n, int k, int relu) {
        dim3 grid(m / BM, n / BN);   // M-block fastest: W slice shared by consecutive blocks
        gemm_f16w<<<grid, 256, 0, stream>>>(A, W, bias, C, m, n, k, relu);
    };
    auto attn = [&](const float* Qp, const float* Kp, const float* Vp, float* Op,
                    const int* tk, int mode) {
        dim3 grid(64 /*B*H*/, S / 128);
        attention_kernel<<<grid, 256, 0, stream>>>(Qp, Kp, Vp, Op, tk, S, S, mode);
    };
    auto ln = [&](const float* X, const float* R, const float* g, const float* bb, float* O) {
        ln_residual_kernel<<<M, 256, 0, stream>>>(X, R, g, bb, O);
    };

    // ---------------- encoder ----------------
    embed_pe_kernel<<<M, 256, 0, stream>>>(src, src_emb, x, S);
    for (int i = 0; i < L; ++i) {
        const float* w  = enc_attn_w + (size_t)i * 4 * WD;
        const float* bb = enc_attn_b + (size_t)i * 4 * Dm;
        gemm(x, w + 0 * WD, bb + 0 * Dm, Qb, M, Dm, Dm, 0);
        gemm(x, w + 1 * WD, bb + 1 * Dm, Kf, M, Dm, Dm, 0);
        gemm(x, w + 2 * WD, bb + 2 * Dm, Vf, M, Dm, Dm, 0);
        attn(Qb, Kf, Vf, aO, src, 0);
        gemm(aO, w + 3 * WD, bb + 3 * Dm, t, M, Dm, Dm, 0);
        ln(x, t, enc_ln_g + (size_t)(i * 2 + 0) * Dm, enc_ln_b + (size_t)(i * 2 + 0) * Dm, x);
        gemm(x,   enc_ff1_w + (size_t)i * Dm * F, enc_ff1_b + (size_t)i * F, ffh, M, F, Dm, 1);
        gemm(ffh, enc_ff2_w + (size_t)i * F * Dm, enc_ff2_b + (size_t)i * Dm, t,  M, Dm, F, 0);
        ln(x, t, enc_ln_g + (size_t)(i * 2 + 1) * Dm, enc_ln_b + (size_t)(i * 2 + 1) * Dm, x);
    }

    // ---------------- decoder ----------------
    embed_pe_kernel<<<M, 256, 0, stream>>>(trg, trg_emb, y, S);
    for (int i = 0; i < L; ++i) {
        // self-attention (causal + query pad)
        const float* wsf = dec_self_w + (size_t)i * 4 * WD;
        const float* bsf = dec_self_b + (size_t)i * 4 * Dm;
        gemm(y, wsf + 0 * WD, bsf + 0 * Dm, Qb, M, Dm, Dm, 0);
        gemm(y, wsf + 1 * WD, bsf + 1 * Dm, Kf, M, Dm, Dm, 0);
        gemm(y, wsf + 2 * WD, bsf + 2 * Dm, Vf, M, Dm, Dm, 0);
        attn(Qb, Kf, Vf, aO, trg, 1);
        gemm(aO, wsf + 3 * WD, bsf + 3 * Dm, t, M, Dm, Dm, 0);
        ln(y, t, dec_ln_g + (size_t)(i * 3 + 0) * Dm, dec_ln_b + (size_t)(i * 3 + 0) * Dm, y);

        // cross-attention (keys/values from encoder output x, src pad mask)
        const float* wc = dec_cross_w + (size_t)i * 4 * WD;
        const float* bc = dec_cross_b + (size_t)i * 4 * Dm;
        gemm(y, wc + 0 * WD, bc + 0 * Dm, Qb, M, Dm, Dm, 0);
        gemm(x, wc + 1 * WD, bc + 1 * Dm, Kf, M, Dm, Dm, 0);
        gemm(x, wc + 2 * WD, bc + 2 * Dm, Vf, M, Dm, Dm, 0);
        attn(Qb, Kf, Vf, aO, src, 0);
        gemm(aO, wc + 3 * WD, bc + 3 * Dm, t, M, Dm, Dm, 0);
        ln(y, t, dec_ln_g + (size_t)(i * 3 + 1) * Dm, dec_ln_b + (size_t)(i * 3 + 1) * Dm, y);

        // feed-forward
        gemm(y,   dec_ff1_w + (size_t)i * Dm * F, dec_ff1_b + (size_t)i * F, ffh, M, F, Dm, 1);
        gemm(ffh, dec_ff2_w + (size_t)i * F * Dm, dec_ff2_b + (size_t)i * Dm, t,  M, Dm, F, 0);
        ln(y, t, dec_ln_g + (size_t)(i * 3 + 2) * Dm, dec_ln_b + (size_t)(i * 3 + 2) * Dm, y);
    }

    // ---------------- output projection: [4096,1024] x [1024,32000] ----------------
    gemm(y, out_w, out_b, (float*)d_out, M, 32000, Dm, 0);
}
